// Tbnn_46402826666691
// MI455X (gfx1250) — matrix operand order repaired and checked by reading, not compiled, not run on hardware
//
#include <hip/hip_runtime.h>

#define H 128
#define TM 128
#define NBASES 10
#define THREADS 256

typedef __attribute__((ext_vector_type(16))) _Float16 v16h;
typedef __attribute__((ext_vector_type(8)))  float    v8f;
typedef __attribute__((ext_vector_type(4)))  float    v4f;

// A-fragment (16x32 f16, row-major in LDS with stride H):
// lane<16 : row = m0+lane,    halves 0-7 = K[kk*32+0..7],  halves 8-15 = K[kk*32+16..23]
// lane>=16: row = m0+lane-16, halves 0-7 = K[kk*32+8..15], halves 8-15 = K[kk*32+24..31]
__device__ __forceinline__ v16h load_a_frag(const _Float16* tile, int m0, int lane, int kk) {
    const int r  = lane & 15;
    const int kh = (lane >> 4) << 3;             // 0 or 8
    const _Float16* p = tile + (m0 + r) * H + (kk << 5) + kh;
    v16h a;
    ((v4f*)&a)[0] = *(const v4f*)(p);            // K+0..7
    ((v4f*)&a)[1] = *(const v4f*)(p + 16);       // K+16..23
    return a;
}

// B-fragment (32x16 f16): lane holds col n0+(lane&15); halves 0-7 = K[kk*32+kh+0..7], halves 8-15 = K[kk*32+16+kh+0..7], kh = 0 or 8
__device__ __forceinline__ v16h load_b_frag(const _Float16* w, int stride, int n0, int lane, int kk) {
    const int kh = (lane >> 4) << 3;
    const _Float16* p = w + ((kk << 5) + kh) * stride + n0 + (lane & 15);
    v16h b;
    #pragma unroll
    for (int s = 0; s < 8; ++s) {
        b[s]     = p[s * stride];
        b[s + 8] = p[(s + 16) * stride];
    }
    return b;
}

__global__ __launch_bounds__(THREADS) void tbnn_wmma_kernel(
    const float* __restrict__ x,  const float* __restrict__ tb,
    const float* __restrict__ W1, const float* __restrict__ b1,
    const float* __restrict__ W2, const float* __restrict__ b2,
    const float* __restrict__ W3, const float* __restrict__ b3,
    const float* __restrict__ W4, const float* __restrict__ b4,
    const float* __restrict__ W5, const float* __restrict__ b5,
    const float* __restrict__ Wc, const float* __restrict__ bc,
    const float* __restrict__ Wg1, const float* __restrict__ bg1,
    const float* __restrict__ Wg2, const float* __restrict__ bg2,
    const float* __restrict__ Wg3, const float* __restrict__ bg3,
    float* __restrict__ out, int n)
{
    extern __shared__ char smem[];
    _Float16* hA   = (_Float16*)smem;            // TM*H f16    (32 KB)
    _Float16* hB   = hA + TM * H;                // TM*H f16    (32 KB)
    _Float16* wbuf = hB + TM * H;                // H*H  f16    (32 KB)
    _Float16* hd   = wbuf + H * H;               // H*16 f16    ( 4 KB)
    float*    tbs  = (float*)(hd + H * 16);      // TM*90 f32   (45 KB) tb tile (async DMA target)
    char*     un   = (char*)(tbs + TM * 90);     // 8 KB region with two disjoint lifetimes:
    float*    xs   = (float*)un;                 //   phase 0-1: x tile (TM*5 f32)
    float*    w1s  = xs + TM * 5;                //   phase 0-1: W1 (5*H f32)
    float*    Pt   = (float*)un;                 //   phase 3-4: head outputs (TM*16 f32)
    float*    bs   = (float*)(un + 8192);        // biases: [0,4H) b2..b5 | [4H,4H+16) head | [4H+16,...) b1

    const int tid  = threadIdx.x;
    const int lane = tid & 31;
    const int wave = tid >> 5;
    const long long tile0 = (long long)blockIdx.x * TM;

    // ---- phase A: kick off async DMA of the tb tile (global -> LDS), overlaps the whole MLP ----
    {
        const unsigned  lds_base = (unsigned)(uintptr_t)tbs;   // LDS aperture: offset lives in addr[31:0]
        const float*    gbase    = tb + tile0 * 90LL;
        const long long gmax     = (long long)n * 360LL - tile0 * 360LL;  // valid bytes in this tile
        for (int i = tid; i < (TM * 90 * 4) / 16; i += THREADS) {         // 2880 x 16B chunks
            const long long gb = (long long)i * 16;
            if (gb + 16 <= gmax) {
                unsigned ldsaddr = lds_base + (unsigned)gb;
                unsigned goff    = (unsigned)gb;
                asm volatile("global_load_async_to_lds_b128 %0, %1, %2 offset:0"
                             :: "v"(ldsaddr), "v"(goff), "s"(gbase) : "memory");
            }
        }
    }

    // ---- phase 0: stage x tile, W1, fused head matrix [Wc | Wg1 Wg2 Wg3 | 0], all biases ----
    for (int i = tid; i < TM * 5; i += THREADS) {
        long long row = tile0 + (i / 5);
        xs[i] = (row < n) ? x[row * 5 + (i % 5)] : 0.f;
    }
    for (int i = tid; i < 5 * H; i += THREADS) w1s[i] = W1[i];
    for (int i = tid; i < H * 16; i += THREADS) {
        int k = i >> 4, j = i & 15;
        float v;
        if      (j < NBASES) v = Wc[k * NBASES + j];
        else if (j == 10)    v = Wg1[k];
        else if (j == 11)    v = Wg2[k];
        else if (j == 12)    v = Wg3[k];
        else                 v = 0.f;
        hd[i] = (_Float16)v;
    }
    for (int i = tid; i < 4 * H; i += THREADS) {
        const float* bp = (i < H) ? b2 : (i < 2 * H) ? b3 : (i < 3 * H) ? b4 : b5;
        bs[i] = bp[i & (H - 1)];
    }
    if (tid < 16) {
        int j = tid;
        bs[4 * H + j] = (j < NBASES) ? bc[j]
                      : (j == 10) ? bg1[0]
                      : (j == 11) ? bg2[0]
                      : (j == 12) ? bg3[0] : 0.f;
    }
    for (int i = tid; i < H; i += THREADS) bs[4 * H + 16 + i] = b1[i];
    __syncthreads();

    // ---- phase 1: layer 1 (K=5, too skinny for WMMA) -> hA in f16 ----
    for (int i = tid; i < TM * H; i += THREADS) {
        int row = i >> 7, col = i & (H - 1);
        float acc = bs[4 * H + 16 + col];
        #pragma unroll
        for (int k = 0; k < 5; ++k) acc = fmaf(xs[row * 5 + k], w1s[k * H + col], acc);
        hA[i] = (_Float16)fmaxf(acc, 0.f);
    }
    __syncthreads();

    // ---- phase 2: layers 2..5 via v_wmma_f32_16x16x32_f16 ----
    const float* Ws[4] = {W2, W3, W4, W5};
    _Float16* cur = hA;
    _Float16* nxt = hB;
    const int m0   = wave << 4;
    const int ncol = lane & 15;
    const int mrow = m0 + ((lane >> 4) << 3);

    for (int l = 0; l < 4; ++l) {
        for (int i = tid; i < H * H; i += THREADS) wbuf[i] = (_Float16)Ws[l][i];
        __syncthreads();

        v16h a0 = load_a_frag(cur, m0, lane, 0);
        v16h a1 = load_a_frag(cur, m0, lane, 1);
        v16h a2 = load_a_frag(cur, m0, lane, 2);
        v16h a3 = load_a_frag(cur, m0, lane, 3);

        #pragma unroll
        for (int nt = 0; nt < 8; ++nt) {
            v8f c;
            #pragma unroll
            for (int r = 0; r < 8; ++r) c[r] = 0.f;
            c = __builtin_amdgcn_wmma_f32_16x16x32_f16(false, a0, false,
                    load_b_frag(wbuf, H, nt * 16, lane, 0), (short)0, c, false, false);
            c = __builtin_amdgcn_wmma_f32_16x16x32_f16(false, a1, false,
                    load_b_frag(wbuf, H, nt * 16, lane, 1), (short)0, c, false, false);
            c = __builtin_amdgcn_wmma_f32_16x16x32_f16(false, a2, false,
                    load_b_frag(wbuf, H, nt * 16, lane, 2), (short)0, c, false, false);
            c = __builtin_amdgcn_wmma_f32_16x16x32_f16(false, a3, false,
                    load_b_frag(wbuf, H, nt * 16, lane, 3), (short)0, c, false, false);

            float bv = bs[l * H + nt * 16 + ncol];
            #pragma unroll
            for (int r = 0; r < 8; ++r) {
                float v = fmaxf(c[r] + bv, 0.f);
                nxt[(mrow + r) * H + nt * 16 + ncol] = (_Float16)v;
            }
        }
        __syncthreads();
        _Float16* t = cur; cur = nxt; nxt = t;
    }

    // ---- phase 3: heads (coeffs | g1 g2 g3) via one WMMA strip per wave ----
    {
        v8f c;
        #pragma unroll
        for (int r = 0; r < 8; ++r) c[r] = 0.f;
        #pragma unroll
        for (int kk = 0; kk < 4; ++kk) {
            v16h a = load_a_frag(cur, m0, lane, kk);
            v16h b = load_b_frag(hd, 16, 0, lane, kk);
            c = __builtin_amdgcn_wmma_f32_16x16x32_f16(false, a, false, b,
                                                       (short)0, c, false, false);
        }
        const int j = lane & 15;
        float hb = bs[4 * H + j];
        #pragma unroll
        for (int r = 0; r < 8; ++r) Pt[(mrow + r) * 16 + j] = c[r] + hb;
    }

    // drain the tb DMA (per-wave ASYNCcnt), then make all LDS writes visible block-wide
    asm volatile("s_wait_asynccnt 0" ::: "memory");
    __syncthreads();

    // ---- phase 4: t0 + sum_t coeffs[t] * tb[p,t,:] entirely from LDS ----
    if (tid < TM) {
        long long p = tile0 + tid;
        if (p < n) {
            const float T01[9] = {-1.f/3.f, 0.f, 0.f, 0.f,  1.f/6.f, 0.f, 0.f, 0.f,  1.f/6.f};
            const float T02[9] = { 1.f/6.f, 0.f, 0.f, 0.f, -1.f/3.f, 0.f, 0.f, 0.f,  1.f/6.f};
            const float T03[9] = { 1.f/6.f, 0.f, 0.f, 0.f,  1.f/6.f, 0.f, 0.f, 0.f, -1.f/3.f};
            float g1 = Pt[tid * 16 + 10], g2 = Pt[tid * 16 + 11], g3 = Pt[tid * 16 + 12];
            float o[9];
            #pragma unroll
            for (int c = 0; c < 9; ++c) o[c] = g1 * T01[c] + g2 * T02[c] + g3 * T03[c];
            const float* tp = tbs + tid * 90;
            #pragma unroll
            for (int t = 0; t < NBASES; ++t) {
                float co = Pt[tid * 16 + t];
                #pragma unroll
                for (int c = 0; c < 9; ++c) o[c] = fmaf(co, tp[t * 9 + c], o[c]);
            }
            #pragma unroll
            for (int c = 0; c < 9; ++c) out[p * 9LL + c] = o[c];
        }
    }
}

extern "C" void kernel_launch(void* const* d_in, const int* in_sizes, int n_in,
                              void* d_out, int out_size, void* d_ws, size_t ws_size,
                              hipStream_t stream) {
    const float* x   = (const float*)d_in[0];
    const float* tb  = (const float*)d_in[1];
    const float* W1  = (const float*)d_in[2];
    const float* b1  = (const float*)d_in[3];
    const float* W2  = (const float*)d_in[4];
    const float* b2  = (const float*)d_in[5];
    const float* W3  = (const float*)d_in[6];
    const float* b3  = (const float*)d_in[7];
    const float* W4  = (const float*)d_in[8];
    const float* b4  = (const float*)d_in[9];
    const float* W5  = (const float*)d_in[10];
    const float* b5  = (const float*)d_in[11];
    const float* Wc  = (const float*)d_in[12];
    const float* bc  = (const float*)d_in[13];
    const float* Wg1 = (const float*)d_in[14];
    const float* bg1 = (const float*)d_in[15];
    const float* Wg2 = (const float*)d_in[16];
    const float* bg2 = (const float*)d_in[17];
    const float* Wg3 = (const float*)d_in[18];
    const float* bg3 = (const float*)d_in[19];

    const int n = in_sizes[0] / 5;
    const int grid = (n + TM - 1) / TM;
    const size_t shmem = (size_t)(2 * TM * H + H * H + H * 16) * sizeof(_Float16)  // hA,hB,wbuf,hd
                       + (size_t)(TM * 90) * sizeof(float)                         // tbs
                       + 8192                                                      // {xs,w1s} / Pt overlay
                       + (size_t)(4 * H + 16 + H) * sizeof(float);                 // biases

    tbnn_wmma_kernel<<<grid, THREADS, shmem, stream>>>(
        x, tb, W1, b1, W2, b2, W3, b3, W4, b4, W5, b5,
        Wc, bc, Wg1, bg1, Wg2, bg2, Wg3, bg3,
        (float*)d_out, n);
}
